// VectorizedConstellationAttention_85710367359291
// MI455X (gfx1250) — matrix operand order repaired and checked by reading, not compiled, not run on hardware
//
#include <hip/hip_runtime.h>
#include <math.h>

typedef __attribute__((ext_vector_type(16))) __bf16 v16bf;
typedef __attribute__((ext_vector_type(8)))  __bf16 v8bf;
typedef __attribute__((ext_vector_type(8)))  float  v8f;

#define D_MODEL  256
#define N_BRANCH 4
#define PH       16
#define KMAX     15
#define REL_HID  64
#define B_SZ     2
#define T_SZ     2048
#define PROJ_N   (N_BRANCH * D_MODEL + D_MODEL)   // 1280
#define SCALE_ACT 1.8137993642342178f             // pi/sqrt(3)

// ---------- WMMA helpers (CDNA5 wave32, 16x16x32 bf16 -> f32) ----------

static __device__ __forceinline__ v8f wmma_bf16(v16bf a, v16bf b, v8f c) {
  return __builtin_amdgcn_wmma_f32_16x16x32_bf16(false, a, false, b, (short)0, c, false, false);
}

static __device__ __forceinline__ v8f splat8(float x) {
  v8f r;
#pragma unroll
  for (int i = 0; i < 8; ++i) r[i] = x;
  return r;
}

// A operand (16x32 bf16, row-major source): lane = row (l&15), k-chunks at
// kbase + hi8 .. +7 and kbase + 16 + hi8 .. +7 where hi8 = (l>=16)*8.
static __device__ __forceinline__ v16bf load_a_frag(const __bf16* tile, int ld, int kbase, int lane) {
  const __bf16* p = tile + (size_t)(lane & 15) * ld + kbase + ((lane >> 4) << 3);
  v8bf lo = *(const v8bf*)(p);
  v8bf hi = *(const v8bf*)(p + 16);
  v16bf a;
#pragma unroll
  for (int i = 0; i < 8; ++i) { a[i] = lo[i]; a[i + 8] = hi[i]; }
  return a;
}

// B operand (32x16 bf16) from an n-major (transposed) buffer Bt[n][k]:
// lane = col (l&15), k-chunks at kbase + hi8 .. +7 and kbase + 16 + hi8 .. +7 where hi8 = (l>=16)*8.
static __device__ __forceinline__ v16bf load_b_frag(const __bf16* tileT, int ld, int kbase, int lane) {
  const __bf16* p = tileT + (size_t)(lane & 15) * ld + kbase + ((lane >> 4) << 3);
  v8bf lo = *(const v8bf*)(p);
  v8bf hi = *(const v8bf*)(p + 16);
  v16bf b;
#pragma unroll
  for (int i = 0; i < 8; ++i) { b[i] = lo[i]; b[i + 8] = hi[i]; }
  return b;
}

// ---------- Kernel 0: pack x and all weights into bf16 (transposed) ----------

#define N_XBF (B_SZ * T_SZ * D_MODEL)   // 1048576
#define N_WT  (PROJ_N * D_MODEL)        // 327680
#define N_W1T (64 * 32)                 // 2048
#define N_W2T (64 * 64)                 // 4096
#define N_WOT (4 * 256 * 256)           // 262144
#define N_PREP (N_XBF + N_WT + N_W1T + N_W2T + N_WOT)

__global__ void k_prep(const float* __restrict__ x, const float* __restrict__ Wi,
                       const float* __restrict__ Wp, const float* __restrict__ W1,
                       const float* __restrict__ W2, const float* __restrict__ Wo,
                       __bf16* __restrict__ xbf, __bf16* __restrict__ Wt,
                       __bf16* __restrict__ W1t, __bf16* __restrict__ W2t,
                       __bf16* __restrict__ Wot) {
  int tid = blockIdx.x * blockDim.x + threadIdx.x;
  if (tid < N_XBF) { xbf[tid] = (__bf16)x[tid]; return; }
  tid -= N_XBF;
  if (tid < N_WT) {  // Wt[n][k] = [Wi|Wp][k][n]
    int n = tid / 256, k = tid % 256;
    float v = (n < 1024) ? Wi[(size_t)k * 1024 + n] : Wp[(size_t)k * 256 + (n - 1024)];
    Wt[tid] = (__bf16)v; return;
  }
  tid -= N_WT;
  if (tid < N_W1T) { // W1t[n][k] (k zero-padded 17->32)
    int n = tid / 32, k = tid % 32;
    float v = (k < 17) ? W1[(size_t)k * 64 + n] : 0.f;
    W1t[tid] = (__bf16)v; return;
  }
  tid -= N_W1T;
  if (tid < N_W2T) { // W2t[n][k]
    int n = tid / 64, k = tid % 64;
    W2t[tid] = (__bf16)W2[(size_t)k * 64 + n]; return;
  }
  tid -= N_W2T;
  if (tid < N_WOT) { // Wot[r][m][d] = Wo[r][d][m]
    int r = tid / 65536, rem = tid % 65536;
    int m = rem / 256, d = rem % 256;
    Wot[tid] = (__bf16)Wo[((size_t)r * 256 + d) * 256 + m];
  }
}

// ---------- Kernel 1: proj = x @ [Wi|Wp]  (16x64 strip per wave, A reused) ----------

__global__ void __launch_bounds__(32) k_gemm_proj(const __bf16* __restrict__ xbf,
                                                  const __bf16* __restrict__ Wt,
                                                  float* __restrict__ proj) {
  int lane = threadIdx.x;
  int mt = blockIdx.x, nt0 = blockIdx.y * 4;
  const __bf16* A = xbf + (size_t)mt * 16 * 256;
  v16bf afr[8];
#pragma unroll
  for (int kk = 0; kk < 8; ++kk) afr[kk] = load_a_frag(A, 256, kk * 32, lane);
  v8f acc[4];
#pragma unroll
  for (int j = 0; j < 4; ++j) acc[j] = splat8(0.f);
#pragma unroll
  for (int kk = 0; kk < 8; ++kk)
#pragma unroll
    for (int j = 0; j < 4; ++j)
      acc[j] = wmma_bf16(afr[kk],
                         load_b_frag(Wt + (size_t)(nt0 + j) * 16 * 256, 256, kk * 32, lane),
                         acc[j]);
  int n = lane & 15, hi = lane >> 4;
#pragma unroll
  for (int j = 0; j < 4; ++j)
#pragma unroll
    for (int v = 0; v < 8; ++v)
      proj[((size_t)(mt * 16 + v + 8 * hi)) * PROJ_N + (nt0 + j) * 16 + n] = acc[j][v];
}

// ---------- Kernel 2: RoPE + split into I (b,r,t,d) and P (b,t,d), f32+bf16 ----------

__global__ void k_rope(const float* __restrict__ proj,
                       float* __restrict__ If32, __bf16* __restrict__ Ibf,
                       float* __restrict__ Pf32, __bf16* __restrict__ Pbf) {
  int tid = blockIdx.x * blockDim.x + threadIdx.x;
  const int total = B_SZ * T_SZ * 5 * 128;
  if (tid >= total) return;
  int i   = tid & 127;
  int c   = (tid >> 7) % 5;
  int tok = tid / (5 * 128);
  int b = tok / T_SZ, t = tok % T_SZ;
  float f  = (float)t * __expf(-(float)i * (9.210340371976184f / 128.f));
  float cs = cosf(f), sn = sinf(f);
  size_t src = (size_t)tok * PROJ_N + c * 256 + 2 * i;
  float x1 = proj[src], x2 = proj[src + 1];
  float y1 = x1 * cs - x2 * sn;
  float y2 = x1 * sn + x2 * cs;
  if (c < 4) {
    size_t dst = (((size_t)(b * 4 + c)) * T_SZ + t) * 256 + 2 * i;
    If32[dst] = y1; If32[dst + 1] = y2;
    Ibf[dst] = (__bf16)y1; Ibf[dst + 1] = (__bf16)y2;
  } else {
    size_t dst = ((size_t)b * T_SZ + t) * 256 + 2 * i;
    Pf32[dst] = y1; Pf32[dst + 1] = y2;
    Pbf[dst] = (__bf16)y1; Pbf[dst + 1] = (__bf16)y2;
  }
}

// ---------- Kernel 3: fused causal logits (WMMA) + streaming top-15 ----------
// A fragments (the I row-tile) hoisted out of the column sweep: the inner loop
// is 8 B-loads + 8 WMMAs per 16-column tile.

__global__ void __launch_bounds__(32) k_topk(const __bf16* __restrict__ Ibf,
                                             const __bf16* __restrict__ Pbf,
                                             float* __restrict__ topv, int* __restrict__ topi) {
  __shared__ float ltile[16][17];
  int lane = threadIdx.x;
  int wid = blockIdx.x;
  int tt = wid % (T_SZ / 16), br = wid / (T_SZ / 16);
  int b = br >> 2;
  int t0 = tt * 16;
  const __bf16* A = Ibf + ((size_t)br * T_SZ + t0) * 256;
  const __bf16* P = Pbf + (size_t)b * T_SZ * 256;

  v16bf afr[8];
#pragma unroll
  for (int kk = 0; kk < 8; ++kk) afr[kk] = load_a_frag(A, 256, kk * 32, lane);

  float vals[KMAX]; int inds[KMAX];
#pragma unroll
  for (int j = 0; j < KMAX; ++j) { vals[j] = -INFINITY; inds[j] = 0; }

  for (int s0 = 0; s0 <= t0; s0 += 16) {
    v8f c = splat8(0.f);
#pragma unroll
    for (int kk = 0; kk < 8; ++kk)
      c = wmma_bf16(afr[kk], load_b_frag(P + (size_t)s0 * 256, 256, kk * 32, lane), c);
    int n = lane & 15, hi = lane >> 4;
#pragma unroll
    for (int v = 0; v < 8; ++v) {
      int m = v + 8 * hi;
      ltile[m][n] = (s0 + n <= t0 + m) ? c[v] * 0.0625f : -INFINITY;   // scale = D^-0.5
    }
    __syncthreads();
    if (lane < 16) {
#pragma unroll
      for (int n2 = 0; n2 < 16; ++n2) {
        float x = ltile[lane][n2];
        if (x > vals[KMAX - 1]) {            // cheap reject vs current 15th
          float cv = x; int ci = s0 + n2;    // sorted-descending insert (ties keep lower s)
#pragma unroll
          for (int j = 0; j < KMAX; ++j) {
            bool sw = cv > vals[j];
            float tv = vals[j]; int ti = inds[j];
            vals[j] = sw ? cv : tv; inds[j] = sw ? ci : ti;
            cv = sw ? tv : cv;      ci = sw ? ti : ci;
          }
        }
      }
    }
    __syncthreads();
  }
  if (lane < 16) {
    size_t row = (size_t)br * T_SZ + t0 + lane;
#pragma unroll
    for (int j = 0; j < KMAX; ++j) { topv[row * KMAX + j] = vals[j]; topi[row * KMAX + j] = inds[j]; }
  }
}

// ---------- Kernel 4: Gram+feat (WMMA), rel MLP (WMMA), softmax, grid-sample -> V ----------

__global__ void __launch_bounds__(32) k_constellation(
    const float* __restrict__ If32, const float* __restrict__ Pf32,
    const float* __restrict__ topv, const int* __restrict__ topi,
    const __bf16* __restrict__ W1t, const float* __restrict__ b1,
    const __bf16* __restrict__ W2t, const float* __restrict__ b2,
    const float* __restrict__ Wc,  const float* __restrict__ bc,
    const float* __restrict__ Wm,  const float* __restrict__ bm,
    const float* __restrict__ pal, __bf16* __restrict__ Vbf) {
  __shared__ __bf16 Mb[16][272];     // rows 0..14: P_sel_n*keep, row 15: I_n
  __shared__ __bf16 relb[16][32];    // rel matrix, K padded 17->32
  __shared__ __bf16 h1b[16][72];
  __shared__ float  h2s[16][68];
  __shared__ float  keepx[16], deltas[16];
  __shared__ int    sidx[16];
  __shared__ float  sw_[16], sx0[16], sy0[16], swx[16], swy[16];

  int lane = threadIdx.x;
  int gid = blockIdx.x;                 // (b*4+r)*T + t
  int t = gid % T_SZ, br = gid / T_SZ, b = br >> 2;
  size_t rowbase = (size_t)br * T_SZ + t;

  if (lane < 16) {
    if (lane < KMAX) {
      float tv = topv[rowbase * KMAX + lane];
      int   ti = topi[rowbase * KMAX + lane];
      float kf = (tv > -INFINITY) ? 1.f : 0.f;
      keepx[lane] = kf; sidx[lane] = ti;
      float d = (float)(t - ti); if (d < 0.f) d = 0.f;
      deltas[lane] = d * (1.f / (float)T_SZ) * kf;
    } else { keepx[lane] = 1.f; sidx[lane] = 0; deltas[lane] = 0.f; }
  }
  __syncthreads();

  // build the 16x256 normalized matrix in LDS (bf16)
  for (int row = 0; row < 16; ++row) {
    const float* src; float mask;
    if (row < KMAX) { src = Pf32 + ((size_t)b * T_SZ + sidx[row]) * 256; mask = keepx[row]; }
    else            { src = If32 + rowbase * 256;                        mask = 1.f; }
    float v[8]; float ss = 0.f;
#pragma unroll
    for (int j = 0; j < 8; ++j) { v[j] = src[lane + 32 * j]; ss += v[j] * v[j]; }
#pragma unroll
    for (int off = 16; off > 0; off >>= 1) ss += __shfl_xor(ss, off, 32);
    float rinv = mask / fmaxf(sqrtf(ss), 1e-12f);
#pragma unroll
    for (int j = 0; j < 8; ++j) Mb[row][lane + 32 * j] = (__bf16)(v[j] * rinv);
  }
  __syncthreads();

  // Gram = M @ M^T : rows give G (15x15), row/col 15 gives feat_a
  v8f g = splat8(0.f);
#pragma unroll
  for (int kk = 0; kk < 256; kk += 32)
    g = wmma_bf16(load_a_frag(&Mb[0][0], 272, kk, lane),
                  load_b_frag(&Mb[0][0], 272, kk, lane), g);

  for (int e = lane; e < 16 * 32; e += 32) ((__bf16*)relb)[e] = (__bf16)0.f;
  __syncthreads();
  {
    int n = lane & 15, hi = lane >> 4;
#pragma unroll
    for (int v = 0; v < 8; ++v) {
      int m = v + 8 * hi;
      float x = fminf(1.f, fmaxf(-1.f, g[v])) * keepx[m] * keepx[n];
      relb[n][m] = (__bf16)x;          // symmetric write: rel[k][0..15] (col 15 = feat_a)
    }
  }
  if (lane < KMAX) relb[lane][16] = (__bf16)deltas[lane];
  __syncthreads();

  // h1 = silu_pi(rel @ W1 + b1)   (one k-step, K padded to 32)
#pragma unroll
  for (int nt = 0; nt < 4; ++nt) {
    v8f h = splat8(b1[nt * 16 + (lane & 15)]);
    h = wmma_bf16(load_a_frag(&relb[0][0], 32, 0, lane),
                  load_b_frag(W1t + (size_t)nt * 16 * 32, 32, 0, lane), h);
    int n = lane & 15, hi = lane >> 4;
#pragma unroll
    for (int v = 0; v < 8; ++v) {
      float x = h[v];
      float a = x / (1.f + __expf(-SCALE_ACT * x));
      h1b[v + 8 * hi][nt * 16 + n] = (__bf16)a;
    }
  }
  __syncthreads();

  // h2 = h1 @ W2 + b2
#pragma unroll
  for (int nt = 0; nt < 4; ++nt) {
    v8f h = splat8(b2[nt * 16 + (lane & 15)]);
#pragma unroll
    for (int kk = 0; kk < 64; kk += 32)
      h = wmma_bf16(load_a_frag(&h1b[0][0], 72, kk, lane),
                    load_b_frag(W2t + (size_t)nt * 16 * 64, 64, kk, lane), h);
    int n = lane & 15, hi = lane >> 4;
#pragma unroll
    for (int v = 0; v < 8; ++v) h2s[v + 8 * hi][nt * 16 + n] = h[v];
  }
  __syncthreads();

  // heads + masked softmax
  float mix = -INFINITY, z0 = 0.f, z1 = 0.f;
  if (lane < KMAX) {
    float c0 = bc[0], c1 = bc[1], mm = bm[0];
    for (int i = 0; i < 64; ++i) {
      float hv = h2s[lane][i];
      c0 += hv * Wc[2 * i]; c1 += hv * Wc[2 * i + 1]; mm += hv * Wm[i];
    }
    z0 = tanhf(c0); z1 = tanhf(c1);
    if (keepx[lane] > 0.f) mix = mm;
  }
  float mx = mix;
#pragma unroll
  for (int off = 16; off > 0; off >>= 1) mx = fmaxf(mx, __shfl_xor(mx, off, 32));
  float e = (mix == -INFINITY) ? 0.f : __expf(mix - mx);
  float se = e;
#pragma unroll
  for (int off = 16; off > 0; off >>= 1) se += __shfl_xor(se, off, 32);
  float w = (se > 0.f) ? e / se : 0.f;
  if (lane < KMAX) {
    float gx = fminf(fmaxf((z0 + 1.f) * 0.5f * (PH - 1), 0.f), (float)(PH - 1));
    float gy = fminf(fmaxf((z1 + 1.f) * 0.5f * (PH - 1), 0.f), (float)(PH - 1));
    float x0 = floorf(gx), y0 = floorf(gy);
    sw_[lane] = w; sx0[lane] = x0; sy0[lane] = y0; swx[lane] = gx - x0; swy[lane] = gy - y0;
  }
  __syncthreads();

  // weighted bilinear palette gather -> V row
  float acc[8];
#pragma unroll
  for (int j = 0; j < 8; ++j) acc[j] = 0.f;
  for (int k = 0; k < KMAX; ++k) {
    float wk = sw_[k];
    if (wk == 0.f) continue;                    // uniform across wave
    int x0 = (int)sx0[k], y0 = (int)sy0[k];
    int x1 = x0 + 1; if (x1 > PH - 1) x1 = PH - 1;
    int y1 = y0 + 1; if (y1 > PH - 1) y1 = PH - 1;
    float wx = swx[k], wy = swy[k];
    float w00 = (1.f - wx) * (1.f - wy) * wk, w01 = wx * (1.f - wy) * wk;
    float w10 = (1.f - wx) * wy * wk,         w11 = wx * wy * wk;
#pragma unroll
    for (int j = 0; j < 8; ++j) {
      const float* pb = pal + (size_t)(lane + 32 * j) * (PH * PH);
      acc[j] += w00 * pb[y0 * PH + x0] + w01 * pb[y0 * PH + x1]
              + w10 * pb[y1 * PH + x0] + w11 * pb[y1 * PH + x1];
    }
  }
#pragma unroll
  for (int j = 0; j < 8; ++j) Vbf[rowbase * 256 + lane + 32 * j] = (__bf16)acc[j];
}

// ---------- Kernel 5: y = mean_r (V_r @ Wo_r)  (16x64 strip, A reused per r) ----------

__global__ void __launch_bounds__(32) k_out(const __bf16* __restrict__ Vbf,
                                            const __bf16* __restrict__ Wot,
                                            float* __restrict__ out) {
  int lane = threadIdx.x;
  int mt = blockIdx.x, nt0 = blockIdx.y * 4;
  int g0 = mt * 16;
  int b = g0 / T_SZ, t0 = g0 % T_SZ;
  v8f acc[4];
#pragma unroll
  for (int j = 0; j < 4; ++j) acc[j] = splat8(0.f);
#pragma unroll
  for (int r = 0; r < 4; ++r) {
    const __bf16* A = Vbf + (((size_t)(b * 4 + r)) * T_SZ + t0) * 256;
    v16bf afr[8];
#pragma unroll
    for (int kk = 0; kk < 8; ++kk) afr[kk] = load_a_frag(A, 256, kk * 32, lane);
#pragma unroll
    for (int kk = 0; kk < 8; ++kk)
#pragma unroll
      for (int j = 0; j < 4; ++j)
        acc[j] = wmma_bf16(afr[kk],
                           load_b_frag(Wot + ((size_t)r * 256 + (nt0 + j) * 16) * 256, 256, kk * 32, lane),
                           acc[j]);
  }
  int n = lane & 15, hi = lane >> 4;
#pragma unroll
  for (int j = 0; j < 4; ++j)
#pragma unroll
    for (int v = 0; v < 8; ++v)
      out[((size_t)g0 + v + 8 * hi) * 256 + (nt0 + j) * 16 + n] = 0.25f * acc[j][v];
}

// ---------- host ----------

extern "C" void kernel_launch(void* const* d_in, const int* in_sizes, int n_in,
                              void* d_out, int out_size, void* d_ws, size_t ws_size,
                              hipStream_t stream) {
  (void)in_sizes; (void)n_in; (void)out_size; (void)ws_size;
  const float* x   = (const float*)d_in[0];
  const float* Wi  = (const float*)d_in[1];
  const float* Wp  = (const float*)d_in[2];
  const float* W1  = (const float*)d_in[3];
  const float* b1  = (const float*)d_in[4];
  const float* W2  = (const float*)d_in[5];
  const float* b2  = (const float*)d_in[6];
  const float* Wc  = (const float*)d_in[7];
  const float* bc  = (const float*)d_in[8];
  const float* Wm  = (const float*)d_in[9];
  const float* bm  = (const float*)d_in[10];
  const float* pal = (const float*)d_in[11];
  const float* Wo  = (const float*)d_in[12];
  float* out = (float*)d_out;

  char* ws = (char*)d_ws;
  size_t off = 0;
  auto carve = [&](size_t bytes) -> void* {
    void* p = ws + off;
    off = (off + bytes + 255) & ~(size_t)255;
    return p;
  };
  __bf16* xbf  = (__bf16*)carve((size_t)N_XBF * 2);
  __bf16* Wt   = (__bf16*)carve((size_t)N_WT * 2);
  __bf16* W1t  = (__bf16*)carve((size_t)N_W1T * 2);
  __bf16* W2t  = (__bf16*)carve((size_t)N_W2T * 2);
  __bf16* Wot  = (__bf16*)carve((size_t)N_WOT * 2);
  float*  proj = (float*) carve((size_t)B_SZ * T_SZ * PROJ_N * 4);
  float*  If32 = (float*) carve((size_t)B_SZ * 4 * T_SZ * 256 * 4);
  __bf16* Ibf  = (__bf16*)carve((size_t)B_SZ * 4 * T_SZ * 256 * 2);
  float*  Pf32 = (float*) carve((size_t)B_SZ * T_SZ * 256 * 4);
  __bf16* Pbf  = (__bf16*)carve((size_t)B_SZ * T_SZ * 256 * 2);
  float*  topv = (float*) carve((size_t)B_SZ * 4 * T_SZ * KMAX * 4);
  int*    topi = (int*)   carve((size_t)B_SZ * 4 * T_SZ * KMAX * 4);
  __bf16* Vbf  = (__bf16*)carve((size_t)B_SZ * 4 * T_SZ * 256 * 2);

  k_prep<<<(N_PREP + 255) / 256, 256, 0, stream>>>(x, Wi, Wp, W1, W2, Wo, xbf, Wt, W1t, W2t, Wot);

  k_gemm_proj<<<dim3(B_SZ * T_SZ / 16, PROJ_N / 64), 32, 0, stream>>>(xbf, Wt, proj);

  const int rope_total = B_SZ * T_SZ * 5 * 128;
  k_rope<<<(rope_total + 255) / 256, 256, 0, stream>>>(proj, If32, Ibf, Pf32, Pbf);

  k_topk<<<B_SZ * 4 * (T_SZ / 16), 32, 0, stream>>>(Ibf, Pbf, topv, topi);

  k_constellation<<<B_SZ * 4 * T_SZ, 32, 0, stream>>>(If32, Pf32, topv, topi,
                                                      W1t, b1, W2t, b2, Wc, bc, Wm, bm,
                                                      pal, Vbf);

  k_out<<<dim3(B_SZ * T_SZ / 16, 256 / 64), 32, 0, stream>>>(Vbf, Wot, out);
}
